// ResidualBlock_85787676770885
// MI455X (gfx1250) — matrix operand order repaired and checked by reading, not compiled, not run on hardware
//
#include <hip/hip_runtime.h>
#include <cstdint>
#include <cstddef>

typedef __attribute__((ext_vector_type(16))) _Float16 v16h;
typedef __attribute__((ext_vector_type(8)))  float    v8f;

#define CH    256
#define TSEQ  4096
#define BATCH 16
#define DIL   8
#define TT    64          // time tile per workgroup
#define LSTRIDE 272       // LDS row stride in halfs: rows 32B-aligned, bank-skewed

// packed-weight layout in d_ws (offsets in halfs)
#define WF_OFF 0          // w_filter: 2 taps * 8 ktiles * 16 mtiles * 512
#define WG_OFF 131072     // w_gate
#define WR_OFF 262144     // w_res:    8 ktiles * 16 mtiles * 512
#define WS_OFF 327680     // w_skip
#define WTOTAL 393216

// ---------------------------------------------------------------------------
// Prepack: fp32 weights -> f16 WMMA A-matrix fragments.
// A 16x32 f16 fragment: lane = hl*16+ln holds A[M=ln, K = hl*16 + p]
// Fragment storage: frag*512 + lane*16 + p  (each lane's 16 halfs = contiguous 32B)
// ---------------------------------------------------------------------------
__global__ void prepack_weights(const float* __restrict__ wf,
                                const float* __restrict__ wg,
                                const float* __restrict__ wr,
                                const float* __restrict__ wsk,
                                _Float16* __restrict__ dst) {
  int idx = blockIdx.x * blockDim.x + threadIdx.x;
  if (idx >= WTOTAL) return;
  const float* src;
  int j;
  bool conv;
  if (idx < WG_OFF)      { src = wf;  j = idx;          conv = true;  }
  else if (idx < WR_OFF) { src = wg;  j = idx - WG_OFF; conv = true;  }
  else if (idx < WS_OFF) { src = wr;  j = idx - WR_OFF; conv = false; }
  else                   { src = wsk; j = idx - WS_OFF; conv = false; }

  int p    = j & 15;
  int lane = (j >> 4) & 31;
  int mt   = (j >> 9) & 15;
  int kt   = (j >> 13) & 7;
  int tap  = conv ? ((j >> 16) & 1) : 0;
  int hl   = lane >> 4;
  int ln   = lane & 15;
  int o    = mt * 16 + ln;                        // output channel (M)
  int kk   = hl * 16 + p;                         // K offset within 32
  int i    = kt * 32 + kk;                        // input channel (K)
  float v  = conv ? src[(o * CH + i) * 2 + tap]   // w[O][I][K], K fastest
                  : src[o * CH + i];              // w[O][I][1]
  dst[idx] = (_Float16)v;
}

// ---------------------------------------------------------------------------
// Main fused kernel: one workgroup = (batch b, 64 time steps), 16 wave32.
// Wave w computes output channels [16w, 16w+16) for all 64 times.
// ---------------------------------------------------------------------------
__global__ __launch_bounds__(512)
void wavenet_block(const float* __restrict__ x,
                   const _Float16* __restrict__ wpack,
                   const float* __restrict__ b_filter,
                   const float* __restrict__ b_gate,
                   const float* __restrict__ b_res,
                   const float* __restrict__ b_skip,
                   float* __restrict__ out_res,
                   float* __restrict__ out_skip) {
  __shared__ __align__(64) _Float16 lds[(TT + DIL) * LSTRIDE]; // 39168 B

  const int tid  = threadIdx.x;
  const int lane = tid & 31;
  const int wv   = tid >> 5;    // wave id = M tile
  const int hl   = lane >> 4;
  const int ln   = lane & 15;
  const int b    = blockIdx.y;
  const int t0   = blockIdx.x * TT;
  const int m0   = wv * 16;
  const float* xb = x + (size_t)b * CH * TSEQ;

  // ---- Phase 1: stage x[b][:, t0-8 .. t0+63] -> LDS f16, time-major ----
  for (int idx = tid; idx < CH * (TT + DIL); idx += 512) {
    int c  = idx / (TT + DIL);
    int t  = idx - c * (TT + DIL);
    int gt = t0 - DIL + t;                       // causal pad: zeros for gt<0
    float v = (gt >= 0) ? xb[(size_t)c * TSEQ + gt] : 0.0f;
    lds[t * LSTRIDE + c] = (_Float16)v;
  }

  // per-lane biases for this wave's channel strip (D-tile row = m0+hl*8+r)
  float bf[8], bg[8], br[8], bs[8];
#pragma unroll
  for (int r = 0; r < 8; ++r) {
    int c = m0 + hl * 8 + r;
    bf[r] = b_filter[c]; bg[r] = b_gate[c];
    br[r] = b_res[c];    bs[r] = b_skip[c];
  }

  __syncthreads();

  // ---- Phase 2: dilated-conv GEMMs (filter + gate), fused K over {tap, C} ----
  v8f accF[4] = {}, accG[4] = {};
#pragma unroll
  for (int tap = 0; tap < 2; ++tap) {
#pragma unroll
    for (int kt = 0; kt < 8; ++kt) {
      const int foff = ((((tap * 8 + kt) * 16) + wv) << 9) + (lane << 4);
      v16h af = *(const v16h*)(wpack + WF_OFF + foff);
      v16h ag = *(const v16h*)(wpack + WG_OFF + foff);
#pragma unroll
      for (int nt = 0; nt < 4; ++nt) {
        // B 32x16 frag: lane group hl holds K = kt*32 + hl*16 + (0..15), N = ln
        // tap0 -> x[t-8] (lds row n), tap1 -> x[t] (lds row n+8)
        const v16h bb = *(const v16h*)&lds[(tap * DIL + nt * 16 + ln) * LSTRIDE
                                           + kt * 32 + hl * 16];
        accF[nt] = __builtin_amdgcn_wmma_f32_16x16x32_f16(
            false, af, false, bb, (short)0, accF[nt], false, false);
        accG[nt] = __builtin_amdgcn_wmma_f32_16x16x32_f16(
            false, ag, false, bb, (short)0, accG[nt], false, false);
      }
    }
  }

  // ---- Phase 3: z = tanh(F+bf) * sigmoid(G+bg), restage z^T into LDS ----
  float z[4][8];
#pragma unroll
  for (int nt = 0; nt < 4; ++nt)
#pragma unroll
    for (int r = 0; r < 8; ++r) {
      float f = accF[nt][r] + bf[r];
      float g = accG[nt][r] + bg[r];
      z[nt][r] = tanhf(f) * (1.0f / (1.0f + __expf(-g)));
    }

  __syncthreads();                         // all waves done reading x tile
#pragma unroll
  for (int nt = 0; nt < 4; ++nt)
#pragma unroll
    for (int r = 0; r < 8; ++r)
      lds[(nt * 16 + ln) * LSTRIDE + (m0 + hl * 8 + r)] = (_Float16)z[nt][r];
  __syncthreads();

  // ---- Phase 4: pointwise GEMMs (residual + skip) over z^T ----
  v8f accR[4] = {}, accS[4] = {};
#pragma unroll
  for (int kt = 0; kt < 8; ++kt) {
    const int foff = (((kt * 16) + wv) << 9) + (lane << 4);
    v16h ar = *(const v16h*)(wpack + WR_OFF + foff);
    v16h as = *(const v16h*)(wpack + WS_OFF + foff);
#pragma unroll
    for (int nt = 0; nt < 4; ++nt) {
      const v16h bb = *(const v16h*)&lds[(nt * 16 + ln) * LSTRIDE
                                         + kt * 32 + hl * 16];
      accR[nt] = __builtin_amdgcn_wmma_f32_16x16x32_f16(
          false, ar, false, bb, (short)0, accR[nt], false, false);
      accS[nt] = __builtin_amdgcn_wmma_f32_16x16x32_f16(
          false, as, false, bb, (short)0, accS[nt], false, false);
    }
  }

  // ---- Phase 5: epilogue: out0 = x + residual + br ; out1 = skip + bs ----
#pragma unroll
  for (int nt = 0; nt < 4; ++nt) {
#pragma unroll
    for (int r = 0; r < 8; ++r) {
      int c = m0 + hl * 8 + r;
      size_t o = ((size_t)b * CH + c) * TSEQ + (size_t)(t0 + nt * 16 + ln);
      out_res[o]  = x[o] + accR[nt][r] + br[r];   // full-precision x re-read
      out_skip[o] = accS[nt][r] + bs[r];
    }
  }
}

// ---------------------------------------------------------------------------
extern "C" void kernel_launch(void* const* d_in, const int* in_sizes, int n_in,
                              void* d_out, int out_size, void* d_ws, size_t ws_size,
                              hipStream_t stream) {
  const float* x        = (const float*)d_in[0];
  const float* w_filter = (const float*)d_in[1];
  const float* b_filter = (const float*)d_in[2];
  const float* w_gate   = (const float*)d_in[3];
  const float* b_gate   = (const float*)d_in[4];
  const float* w_res    = (const float*)d_in[5];
  const float* b_res    = (const float*)d_in[6];
  const float* w_skip   = (const float*)d_in[7];
  const float* b_skip   = (const float*)d_in[8];

  float* out_res  = (float*)d_out;
  float* out_skip = out_res + (size_t)BATCH * CH * TSEQ;
  _Float16* wpack = (_Float16*)d_ws;   // 768 KB of packed f16 weights

  prepack_weights<<<dim3((WTOTAL + 255) / 256), dim3(256), 0, stream>>>(
      w_filter, w_gate, w_res, w_skip, wpack);

  dim3 grid(TSEQ / TT, BATCH);         // 64 x 16 = 1024 workgroups
  wavenet_block<<<grid, dim3(512), 0, stream>>>(
      x, wpack, b_filter, b_gate, b_res, b_skip, out_res, out_skip);
}
